// GlobalAttention_18159121727873
// MI455X (gfx1250) — matrix operand order repaired and checked by reading, not compiled, not run on hardware
//
#include <hip/hip_runtime.h>

// ---------------------------------------------------------------------------
// CDNA5 (gfx1250) global-attention kernel.  wave32, WMMA bf16 path.
// Fragment-swizzled weights + packed LDS traffic.
// ---------------------------------------------------------------------------

typedef __bf16 bf16;
typedef __bf16 v16bf __attribute__((ext_vector_type(16)));
typedef __bf16 bf8v  __attribute__((ext_vector_type(8)));
typedef __bf16 bf4v  __attribute__((ext_vector_type(4)));
typedef float  v8f   __attribute__((ext_vector_type(8)));

#define S_LEN    2048
#define CE       64
#define CH       8
#define NH       8
#define TILE_S   128
#define NTILES   (S_LEN / TILE_S)
#define NTHREADS 256

// LDS layout (bytes from base of dynamic shared)
#define OFF_KV    0        // [16 cols][2048 s] bf16 (cols 0..7=K, 8..15=V)  65536 B
#define OFF_MT    65536    // [128][64]  bf16  staged m tile / gated tile    16384 B
#define OFF_WG    81920    // Wg fragment-swizzled bf16 (8 frags x 1KB)       8192 B
#define OFF_WO    90112    // Wo fragment-swizzled bf16 (8 frags x 1KB)       8192 B
#define OFF_WKV   98304    // [Wk|Wv] fragment-swizzled bf16 (2 frags)        2048 B
#define OFF_MASK  100352   // [2048] f32                                      8192 B
#define OFF_COL   108544   // [64]   f32  masked column sums                   256 B
#define OFF_Q     108800   // [64]   f32  scaled query                         256 B
#define OFF_O     109056   // [64]   f32  attention output o[h*8+c]            256 B
#define OFF_MSUM  109312   // [1]    f32  mask sum                              16 B
#define SMEM_BYTES 109328

__device__ __forceinline__ v8f wmma_bf16f32(v16bf a, v16bf b, v8f c) {
  // D = A(16x32 bf16) * B(32x16 bf16) + C(16x16 f32)
  return __builtin_amdgcn_wmma_f32_16x16x32_bf16(
      /*neg_a=*/false, a, /*neg_b=*/false, b,
      /*c_mod=*/(short)0, c, /*reuse_a=*/false, /*reuse_b=*/false);
}

// A-fragment (16x32) from row-major bf16 LDS tile.
// A layout (contiguous k, matches B staging): lanes 0-15 hold K 0-15, lanes 16-31 hold K 16-31.
__device__ __forceinline__ v16bf load_a_frag(const bf16* mt, int rowBase, int kb, int lane) {
  const int lr = lane & 15;
  const int ho = (lane >> 4) << 4;                 // 0 or 16
  const bf16* p = mt + (rowBase + lr) * CE + kb + ho;
  union { v16bf v; bf8v h[2]; } u;
  u.h[0] = *(const bf8v*)p;
  u.h[1] = *(const bf8v*)(p + 8);
  return u.v;
}

// B-fragment from fragment-swizzled LDS: per fragment, lane-major, each lane's
// 16 bf16 contiguous -> two ds_load_b128, no repacking.
__device__ __forceinline__ v16bf load_b_frag(const bf16* fragBuf, int fid, int lane) {
  const bf16* p = fragBuf + (((fid << 5) + lane) << 4);
  union { v16bf v; bf8v h[2]; } u;
  u.h[0] = *(const bf8v*)p;
  u.h[1] = *(const bf8v*)(p + 8);
  return u.v;
}

// Stage row-major [64 x ncols] f32 weight into fragment-swizzled bf16 LDS.
// Fragment fid = ks*(ncols/16) + j holds B(32x16) for K=32*ks..+31, N=16*j..+15:
// element (lane, e): K = 32*ks + 16*(lane>>4) + e, N = 16*j + (lane&15).
__device__ __forceinline__ void stage_frags(bf16* dst, const float* __restrict__ W,
                                            int ncols, int t) {
  const int nchunks = ncols >> 4;
  for (int idx = t; idx < CE * ncols; idx += NTHREADS) {
    const int e   = idx & 15;
    const int ln  = (idx >> 4) & 31;
    const int fid = idx >> 9;
    const int ks  = fid / nchunks;
    const int j   = fid - ks * nchunks;
    const int k   = (ks << 5) + ((ln >> 4) << 4) + e;
    const int col = (j << 4) + (ln & 15);
    dst[idx] = (bf16)W[k * ncols + col];
  }
}

__global__ __launch_bounds__(NTHREADS) void ga_kernel(
    const float* __restrict__ mG,  const float* __restrict__ maskG,
    const float* __restrict__ WqG, const float* __restrict__ WkG,
    const float* __restrict__ WvG, const float* __restrict__ WgG,
    const float* __restrict__ bgG, const float* __restrict__ WoG,
    const float* __restrict__ boG, float* __restrict__ outG) {
  extern __shared__ char smem[];
  bf16*  kvL   = (bf16*)(smem + OFF_KV);
  bf16*  mtL   = (bf16*)(smem + OFF_MT);
  bf16*  wgL   = (bf16*)(smem + OFF_WG);
  bf16*  woL   = (bf16*)(smem + OFF_WO);
  bf16*  wkvL  = (bf16*)(smem + OFF_WKV);
  float* maskL = (float*)(smem + OFF_MASK);
  float* colL  = (float*)(smem + OFF_COL);
  float* qL    = (float*)(smem + OFF_Q);
  float* oL    = (float*)(smem + OFF_O);
  float* msumL = (float*)(smem + OFF_MSUM);

  const int t    = threadIdx.x;
  const int lane = t & 31;
  const int w    = t >> 5;                 // wave id 0..7
  const int n    = blockIdx.x;             // one block per (b, n) row
  const size_t mBase = (size_t)n * S_LEN * CE;

  if (t < CE) colL[t] = 0.f;
  if (t == CE) msumL[0] = 0.f;
  __syncthreads();

  // ---- stage mask (f32) + mask sum -----------------------------------------
  {
    float ls = 0.f;
#pragma unroll
    for (int i = 0; i < S_LEN / NTHREADS; ++i) {
      int s = t + i * NTHREADS;
      float v = maskG[(size_t)n * S_LEN + s];
      maskL[s] = v;
      ls += v;
    }
    atomicAdd(msumL, ls);
  }
  // ---- stage weights fragment-swizzled (bf16) ------------------------------
  stage_frags(wgL, WgG, CE, t);
  stage_frags(woL, WoG, CE, t);
  for (int idx = t; idx < CE * 16; idx += NTHREADS) {
    const int e  = idx & 15;
    const int ln = (idx >> 4) & 31;
    const int ks = idx >> 9;                          // nchunks == 1
    const int k  = (ks << 5) + ((ln >> 4) << 4) + e;
    const int col = ln & 15;
    const float v = (col < CH) ? WkG[k * CH + col] : WvG[k * CH + (col - CH)];
    wkvL[idx] = (bf16)v;
  }
  __syncthreads();

  const int rRow = t >> 4;                 // 0..15 : row-in-16-group for staging
  const int c4   = (t & 15) * 4;           // fixed 4-channel group for staging
  const int lr   = lane & 15;
  const int mo   = (lane >> 4) << 3;       // C/D: lanes>=16 hold rows M=8..15

  // ============ Phase A: K|V via WMMA + masked column sums ==================
  float csum[4] = {0.f, 0.f, 0.f, 0.f};
  for (int tile = 0; tile < NTILES; ++tile) {
    const int s0 = tile * TILE_S;
    if (tile + 1 < NTILES)
      __builtin_prefetch(mG + mBase + (size_t)(s0 + TILE_S) * CE, 0, 0);
#pragma unroll
    for (int i = 0; i < TILE_S / 16; ++i) {
      int row = i * 16 + rRow;
      const float4 v = *(const float4*)(mG + mBase + (size_t)(s0 + row) * CE + c4);
      float mv = maskL[s0 + row];
      csum[0] += mv * v.x; csum[1] += mv * v.y;
      csum[2] += mv * v.z; csum[3] += mv * v.w;
      bf4v pk;
      pk[0] = (bf16)v.x; pk[1] = (bf16)v.y; pk[2] = (bf16)v.z; pk[3] = (bf16)v.w;
      *(bf4v*)(mtL + row * CE + c4) = pk;
    }
    __syncthreads();
    {
      v16bf a0 = load_a_frag(mtL, 16 * w, 0,  lane);
      v16bf a1 = load_a_frag(mtL, 16 * w, 32, lane);
      v8f acc = {0.f,0.f,0.f,0.f,0.f,0.f,0.f,0.f};
      acc = wmma_bf16f32(a0, load_b_frag(wkvL, 0, lane), acc);
      acc = wmma_bf16f32(a1, load_b_frag(wkvL, 1, lane), acc);
      // kv column-major: one packed 16B store per lane (8 consecutive s)
      const int sA = s0 + 16 * w + mo;
      bf8v pk;
#pragma unroll
      for (int r = 0; r < 8; ++r) pk[r] = (bf16)acc[r];
      *(bf8v*)(kvL + lr * S_LEN + sA) = pk;
    }
    __syncthreads();
  }
#pragma unroll
  for (int qd = 0; qd < 4; ++qd) atomicAdd(&colL[c4 + qd], csum[qd]);
  __syncthreads();

  // ============ Phase B: query (wave 0, fp32) ===============================
  if (w == 0) {
    const float inv = 1.f / (msumL[0] + 1e-10f);
#pragma unroll
    for (int half = 0; half < 2; ++half) {
      int hc = lane + 32 * half;
      float acc = 0.f;
#pragma unroll
      for (int c = 0; c < CE; ++c) acc += (colL[c] * inv) * WqG[c * CE + hc];
      qL[hc] = acc * 0.35355339059327373f;   // 1/sqrt(C_H)
    }
  }
  __syncthreads();

  // ============ Phase C: per-head online softmax (wave w = head w) ==========
  {
    float qh[8];
#pragma unroll
    for (int c = 0; c < 8; ++c) qh[c] = qL[w * 8 + c];
    float mx = -3.0e38f, sm = 0.f;
    float ac[8] = {0,0,0,0,0,0,0,0};
#pragma unroll 1
    for (int it = 0; it < S_LEN / 256; ++it) {     // 8 iterations, 8 s per lane
      const int sB = it * 256 + lane * 8;
      bf8v k8[8], v8[8];
#pragma unroll
      for (int c = 0; c < 8; ++c) {
        k8[c] = *(const bf8v*)(kvL + c * S_LEN + sB);
        v8[c] = *(const bf8v*)(kvL + (8 + c) * S_LEN + sB);
      }
      const float4 mk0 = *(const float4*)(maskL + sB);
      const float4 mk1 = *(const float4*)(maskL + sB + 4);
      float mk[8] = {mk0.x, mk0.y, mk0.z, mk0.w, mk1.x, mk1.y, mk1.z, mk1.w};
      float lg[8];
#pragma unroll
      for (int u = 0; u < 8; ++u) {
        float l = (mk[u] - 1.f) * 1e9f;
#pragma unroll
        for (int c = 0; c < 8; ++c) l += qh[c] * (float)k8[c][u];
        lg[u] = l;
      }
      float lm = lg[0];
#pragma unroll
      for (int u = 1; u < 8; ++u) lm = fmaxf(lm, lg[u]);
      const float nm = fmaxf(mx, lm);
      const float e0 = __expf(mx - nm);
      float p[8], ps = 0.f;
#pragma unroll
      for (int u = 0; u < 8; ++u) { p[u] = __expf(lg[u] - nm); ps += p[u]; }
      sm = sm * e0 + ps;
#pragma unroll
      for (int c = 0; c < 8; ++c) {
        float s8 = 0.f;
#pragma unroll
        for (int u = 0; u < 8; ++u) s8 += p[u] * (float)v8[c][u];
        ac[c] = ac[c] * e0 + s8;
      }
      mx = nm;
    }
#pragma unroll
    for (int off = 16; off > 0; off >>= 1) {
      float omx = __shfl_xor(mx, off);
      float osm = __shfl_xor(sm, off);
      float nm  = fmaxf(mx, omx);
      float e0  = __expf(mx - nm), e1 = __expf(omx - nm);
      sm = sm * e0 + osm * e1;
#pragma unroll
      for (int c = 0; c < 8; ++c) {
        float oa = __shfl_xor(ac[c], off);
        ac[c] = ac[c] * e0 + oa * e1;
      }
      mx = nm;
    }
    if (lane == 0) {
      float inv = 1.f / sm;
#pragma unroll
      for (int c = 0; c < 8; ++c) oL[w * 8 + c] = ac[c] * inv;
    }
  }
  __syncthreads();

  // ============ Phase D: gate GEMM + output GEMM ============================
  for (int tile = 0; tile < NTILES; ++tile) {
    const int s0 = tile * TILE_S;
    if (tile + 1 < NTILES)
      __builtin_prefetch(mG + mBase + (size_t)(s0 + TILE_S) * CE, 0, 0);
#pragma unroll
    for (int i = 0; i < TILE_S / 16; ++i) {
      int row = i * 16 + rRow;
      const float4 v = *(const float4*)(mG + mBase + (size_t)(s0 + row) * CE + c4);
      bf4v pk;
      pk[0] = (bf16)v.x; pk[1] = (bf16)v.y; pk[2] = (bf16)v.z; pk[3] = (bf16)v.w;
      *(bf4v*)(mtL + row * CE + c4) = pk;
    }
    __syncthreads();

    // g_pre = m_tile @ Wg   (per wave: its 16-row block, full N=64)
    v16bf a0 = load_a_frag(mtL, 16 * w, 0,  lane);
    v16bf a1 = load_a_frag(mtL, 16 * w, 32, lane);
    v8f z = {0.f,0.f,0.f,0.f,0.f,0.f,0.f,0.f};
    v8f gacc[4];
#pragma unroll
    for (int j = 0; j < 4; ++j) {
      gacc[j] = z;
      gacc[j] = wmma_bf16f32(a0, load_b_frag(wgL, j,     lane), gacc[j]);
      gacc[j] = wmma_bf16f32(a1, load_b_frag(wgL, 4 + j, lane), gacc[j]);
    }
    // t = sigmoid(g_pre + bg) * o[hc], written back into mtL as bf16.
    // Only this wave touches its 16-row block; LDS ops within a wave are
    // in program order, so no barrier is needed between read and write.
#pragma unroll
    for (int j = 0; j < 4; ++j) {
      const int N   = 16 * j + lr;
      const float bgv = bgG[N];
      const float ov  = oL[N];
#pragma unroll
      for (int r = 0; r < 8; ++r) {
        float gp = gacc[j][r] + bgv;
        float g  = 1.f / (1.f + __expf(-gp));
        mtL[(16 * w + mo + r) * CE + N] = (bf16)(g * ov);
      }
    }
    // out = t @ Wo + bo
    v16bf t0 = load_a_frag(mtL, 16 * w, 0,  lane);
    v16bf t1 = load_a_frag(mtL, 16 * w, 32, lane);
#pragma unroll
    for (int j = 0; j < 4; ++j) {
      v8f oacc = z;
      oacc = wmma_bf16f32(t0, load_b_frag(woL, j,     lane), oacc);
      oacc = wmma_bf16f32(t1, load_b_frag(woL, 4 + j, lane), oacc);
      const int N   = 16 * j + lr;
      const float bov = boG[N];
      float* orow = outG + mBase + (size_t)(s0 + 16 * w + mo) * CE + N;
#pragma unroll
      for (int r = 0; r < 8; ++r) orow[r * CE] = oacc[r] + bov;
    }
    __syncthreads();   // before next tile overwrites mtL
  }
}

extern "C" void kernel_launch(void* const* d_in, const int* in_sizes, int n_in,
                              void* d_out, int out_size, void* d_ws, size_t ws_size,
                              hipStream_t stream) {
  (void)n_in; (void)out_size; (void)d_ws; (void)ws_size;
  const float* m    = (const float*)d_in[0];
  const float* mask = (const float*)d_in[1];
  const float* Wq   = (const float*)d_in[2];
  const float* Wk   = (const float*)d_in[3];
  const float* Wv   = (const float*)d_in[4];
  const float* Wg   = (const float*)d_in[5];
  const float* bg   = (const float*)d_in[6];
  const float* Wo   = (const float*)d_in[7];
  const float* bo   = (const float*)d_in[8];
  float* out = (float*)d_out;
  const int nBlocks = in_sizes[1] / S_LEN;   // B*N rows, one block each
  ga_kernel<<<nBlocks, NTHREADS, SMEM_BYTES, stream>>>(
      m, mask, Wq, Wk, Wv, Wg, bg, Wo, bo, out);
}
